// BahdanauAttention_24000277250165
// MI455X (gfx1250) — matrix operand order repaired and checked by reading, not compiled, not run on hardware
//
#include <hip/hip_runtime.h>

// Problem constants (match reference)
#define BB 8
#define TT 256
#define SS 256
#define HH 512
#define BT (BB * TT)   // 2048 rows (query side)
#define BS (BB * SS)   // 2048 rows (encoder side)

typedef __attribute__((ext_vector_type(16))) __bf16 v16bf;
typedef __attribute__((ext_vector_type(8)))  __bf16 v8bf;
typedef __attribute__((ext_vector_type(4)))  __bf16 v4bf;
typedef __attribute__((ext_vector_type(8)))  float  v8f;

// ---------------------------------------------------------------------------
// Fast tanh via native v_exp_f32: tanh(x) = sign(x)*(1-e^{-2|x|})/(1+e^{-2|x|})
// ---------------------------------------------------------------------------
__device__ __forceinline__ float fast_tanh(float x) {
    float ax = __builtin_fabsf(x);
    float t  = __expf(-2.0f * ax);
    float r  = (1.0f - t) * __builtin_amdgcn_rcpf(1.0f + t);
    return __builtin_copysignf(r, x);
}

// ---------------------------------------------------------------------------
// Elementwise f32 -> bf16 pack (float4 -> 4x bf16 per thread). n % 1024 == 0.
// ---------------------------------------------------------------------------
__global__ __launch_bounds__(256)
void pack_bf16_kernel(const float* __restrict__ src, __bf16* __restrict__ dst, int n) {
    int i = (blockIdx.x * 256 + threadIdx.x) * 4;
    if (i < n) {
        float4 f = *(const float4*)(src + i);
        v4bf o;
        o[0] = (__bf16)f.x; o[1] = (__bf16)f.y;
        o[2] = (__bf16)f.z; o[3] = (__bf16)f.w;
        *(v4bf*)(dst + i) = o;
    }
}

// ---------------------------------------------------------------------------
// bf16 WMMA fragment loaders (packed bf16 source, pure b128 loads).
// A: 16x32 (MxK). Lane l: row m=l&15, kb=(l>>4)*16.
//   a[0..7]  = K = kb..kb+7      (contiguous, 16B)
//   a[8..15] = K = kb+8..kb+15   (contiguous, 16B)
// ---------------------------------------------------------------------------
__device__ __forceinline__ v16bf load_a_frag(const __bf16* __restrict__ A, int lda,
                                             int m0, int k0, int lane) {
    const int m  = m0 + (lane & 15);
    const int kb = (lane >> 4) * 16;
    const __bf16* p = A + (size_t)m * lda + k0 + kb;
    v8bf lo = *(const v8bf*)p;
    v8bf hi = *(const v8bf*)(p + 8);
    v16bf a;
#pragma unroll
    for (int j = 0; j < 8; ++j) { a[j] = lo[j]; a[8 + j] = hi[j]; }
    return a;
}

// B: 32x16 (KxN) from row-major weight W (N x K): B[k][n] = W[n][k].
// Lane l: column n=l&15, kb2=(l>>4)*16; b[0..15] = K = kb2..kb2+15 (32B).
__device__ __forceinline__ v16bf load_b_frag(const __bf16* __restrict__ W, int ldw,
                                             int n, int k0, int lane) {
    const int kb = (lane >> 4) * 16;
    const __bf16* p = W + (size_t)n * ldw + k0 + kb;
    v8bf lo = *(const v8bf*)p;
    v8bf hi = *(const v8bf*)(p + 8);
    v16bf b;
#pragma unroll
    for (int j = 0; j < 8; ++j) { b[j] = lo[j]; b[8 + j] = hi[j]; }
    return b;
}

__device__ __forceinline__ void store_tile_f32(float* __restrict__ Y, int N,
                                               int m0, int n0, int lane, v8f c) {
    const int n = n0 + (lane & 15);
    const int mbase = m0 + 8 * (lane >> 4);
#pragma unroll
    for (int r = 0; r < 8; ++r)
        Y[(size_t)(mbase + r) * N + n] = c[r];
}

// ---------------------------------------------------------------------------
// GEMM: Y(M x N, f32) = X(M x K, bf16) @ W(N x K, bf16)^T
// block = 128 threads (4 waves). Each wave computes a 16(M) x 64(N) strip:
// one A fragment feeds 4 WMMAs per k-step. grid = (M/16, N/256).
// Exact tiling -> EXEC all-1s at every WMMA.
// ---------------------------------------------------------------------------
__global__ __launch_bounds__(128)
void gemm_bf16_kernel(const __bf16* __restrict__ X, const __bf16* __restrict__ W,
                      float* __restrict__ Y, int N, int K) {
    const int lane = threadIdx.x & 31;
    const int wave = threadIdx.x >> 5;
    const int m0 = blockIdx.x * 16;
    const int n0 = blockIdx.y * 256 + wave * 64;
    const int nl = lane & 15;

    v8f c0 = {}, c1 = {}, c2 = {}, c3 = {};
    for (int k0 = 0; k0 < K; k0 += 32) {
        v16bf a  = load_a_frag(X, K, m0, k0, lane);
        v16bf b0 = load_b_frag(W, K, n0 + nl,      k0, lane);
        v16bf b1 = load_b_frag(W, K, n0 + 16 + nl, k0, lane);
        v16bf b2 = load_b_frag(W, K, n0 + 32 + nl, k0, lane);
        v16bf b3 = load_b_frag(W, K, n0 + 48 + nl, k0, lane);
        c0 = __builtin_amdgcn_wmma_f32_16x16x32_bf16(false, a, false, b0, (short)0, c0, false, false);
        c1 = __builtin_amdgcn_wmma_f32_16x16x32_bf16(false, a, false, b1, (short)0, c1, false, false);
        c2 = __builtin_amdgcn_wmma_f32_16x16x32_bf16(false, a, false, b2, (short)0, c2, false, false);
        c3 = __builtin_amdgcn_wmma_f32_16x16x32_bf16(false, a, false, b3, (short)0, c3, false, false);
    }
    store_tile_f32(Y, N, m0, n0,      lane, c0);
    store_tile_f32(Y, N, m0, n0 + 16, lane, c1);
    store_tile_f32(Y, N, m0, n0 + 32, lane, c2);
    store_tile_f32(Y, N, m0, n0 + 48, lane, c3);
}

// ---------------------------------------------------------------------------
// Fused energies -> mask -> softmax -> context. One block per (b,t).
//   e[s]   = sum_h v[h] * tanh(Wsq[b,t,h] + Whe[b,s,h]);  s >= len -> -inf
//   w      = softmax(e);  ctx_bf[b,t,h] = (bf16) sum_s w[s]*enc[b,s,h]
// ---------------------------------------------------------------------------
__global__ __launch_bounds__(256)
void energies_softmax_ctx_kernel(const float* __restrict__ Wsq,
                                 const float* __restrict__ Whe,
                                 const float* __restrict__ enc,
                                 const float* __restrict__ v,
                                 const int* __restrict__ lens,
                                 __bf16* __restrict__ ctx_bf) {
    const int b = blockIdx.x / TT;
    const int t = blockIdx.x % TT;
    const int tid  = threadIdx.x;
    const int lane = tid & 31;
    const int wave = tid >> 5;

    __shared__ float sq[HH];
    __shared__ float sv[HH];
    __shared__ float se[SS];
    __shared__ float red[256];

    const float* qrow = Wsq + ((size_t)b * TT + t) * HH;
    for (int h = tid; h < HH; h += 256) {
        sq[h] = qrow[h];
        sv[h] = v[h];
    }
    __syncthreads();

    const int len = lens[b];

    // ---- energies (transcendental-bound: native v_exp path) ----
    for (int s = wave; s < SS; s += 8) {
        const float* wh = Whe + ((size_t)b * SS + s) * HH;
        if (s + 8 < SS)
            __builtin_prefetch(Whe + ((size_t)b * SS + s + 8) * HH + lane * 4, 0, 3);
        float acc = 0.0f;
#pragma unroll 4
        for (int h = lane; h < HH; h += 32)
            acc += sv[h] * fast_tanh(sq[h] + wh[h]);
#pragma unroll
        for (int off = 16; off > 0; off >>= 1)
            acc += __shfl_down(acc, off, 32);
        if (lane == 0)
            se[s] = (s < len) ? acc : -__builtin_inff();
    }
    __syncthreads();

    // ---- softmax over S=256 (one element per thread) ----
    float e = se[tid];
    red[tid] = e;
    __syncthreads();
#pragma unroll
    for (int st = 128; st > 0; st >>= 1) {
        if (tid < st) red[tid] = fmaxf(red[tid], red[tid + st]);
        __syncthreads();
    }
    const float mx = red[0];
    __syncthreads();
    float p = __expf(e - mx);            // masked: exp(-inf) = 0
    red[tid] = p;
    __syncthreads();
#pragma unroll
    for (int st = 128; st > 0; st >>= 1) {
        if (tid < st) red[tid] += red[tid + st];
        __syncthreads();
    }
    const float inv_sum = __builtin_amdgcn_rcpf(red[0]);
    se[tid] = p * inv_sum;
    __syncthreads();

    // ---- context (coalesced over h), stored as bf16 for the output GEMM ----
    __bf16* crow = ctx_bf + ((size_t)b * TT + t) * HH;
    for (int h = tid; h < HH; h += 256) {
        const float* ecol = enc + (size_t)b * SS * HH + h;
        float acc = 0.0f;
#pragma unroll 8
        for (int s = 0; s < SS; ++s)
            acc += se[s] * ecol[(size_t)s * HH];
        crow[h] = (__bf16)acc;
    }
}

// ---------------------------------------------------------------------------
// Output GEMM: out = tanh([ctx | query] @ W_out^T); M=2048, N=512, K=1024.
// A fragments come from ctx_bf (k<H) or q_bf (k>=H); k0 step 32 never
// straddles the boundary (H % 32 == 0). Same 16x64 strip per wave.
// ---------------------------------------------------------------------------
__global__ __launch_bounds__(128)
void gemm_out_kernel(const __bf16* __restrict__ Ctx, const __bf16* __restrict__ Q,
                     const __bf16* __restrict__ Wout, float* __restrict__ Y) {
    const int lane = threadIdx.x & 31;
    const int wave = threadIdx.x >> 5;
    const int m0 = blockIdx.x * 16;
    const int n0 = blockIdx.y * 256 + wave * 64;
    const int nl = lane & 15;

    v8f c0 = {}, c1 = {}, c2 = {}, c3 = {};
    for (int k0 = 0; k0 < 2 * HH; k0 += 32) {
        const __bf16* src = (k0 < HH) ? Ctx : Q;
        const int kk = (k0 < HH) ? k0 : (k0 - HH);
        v16bf a  = load_a_frag(src, HH, m0, kk, lane);
        v16bf b0 = load_b_frag(Wout, 2 * HH, n0 + nl,      k0, lane);
        v16bf b1 = load_b_frag(Wout, 2 * HH, n0 + 16 + nl, k0, lane);
        v16bf b2 = load_b_frag(Wout, 2 * HH, n0 + 32 + nl, k0, lane);
        v16bf b3 = load_b_frag(Wout, 2 * HH, n0 + 48 + nl, k0, lane);
        c0 = __builtin_amdgcn_wmma_f32_16x16x32_bf16(false, a, false, b0, (short)0, c0, false, false);
        c1 = __builtin_amdgcn_wmma_f32_16x16x32_bf16(false, a, false, b1, (short)0, c1, false, false);
        c2 = __builtin_amdgcn_wmma_f32_16x16x32_bf16(false, a, false, b2, (short)0, c2, false, false);
        c3 = __builtin_amdgcn_wmma_f32_16x16x32_bf16(false, a, false, b3, (short)0, c3, false, false);
    }
    // tanh + store
    const int n = n0 + nl;
    const int mbase = m0 + 8 * (lane >> 4);
#pragma unroll
    for (int r = 0; r < 8; ++r) {
        Y[(size_t)(mbase + r) * HH + (n)]      = fast_tanh(c0[r]);
        Y[(size_t)(mbase + r) * HH + (n + 16)] = fast_tanh(c1[r]);
        Y[(size_t)(mbase + r) * HH + (n + 32)] = fast_tanh(c2[r]);
        Y[(size_t)(mbase + r) * HH + (n + 48)] = fast_tanh(c3[r]);
    }
}

// ---------------------------------------------------------------------------
extern "C" void kernel_launch(void* const* d_in, const int* in_sizes, int n_in,
                              void* d_out, int out_size, void* d_ws, size_t ws_size,
                              hipStream_t stream) {
    const float* query = (const float*)d_in[0];   // (B,T,H)
    const float* enc   = (const float*)d_in[1];   // (B,S,H)
    const int*   lens  = (const int*)d_in[2];     // (B,)
    const float* W_s   = (const float*)d_in[3];   // (H,H)
    const float* W_h   = (const float*)d_in[4];   // (H,H)
    const float* v     = (const float*)d_in[5];   // (H,)
    const float* W_out = (const float*)d_in[6];   // (H,2H)
    float* out = (float*)d_out;                   // (B,T,H)

    // ---- workspace layout ----
    char* ws = (char*)d_ws;
    float*  wsq    = (float*)ws;                      ws += (size_t)BT * HH * 4;  // 4 MB
    float*  whe    = (float*)ws;                      ws += (size_t)BS * HH * 4;  // 4 MB
    __bf16* q_bf   = (__bf16*)ws;                     ws += (size_t)BT * HH * 2;  // 2 MB
    __bf16* e_bf   = (__bf16*)ws;                     ws += (size_t)BS * HH * 2;  // 2 MB
    __bf16* ctx_bf = (__bf16*)ws;                     ws += (size_t)BT * HH * 2;  // 2 MB
    __bf16* Ws_bf  = (__bf16*)ws;                     ws += (size_t)HH * HH * 2;  // 512 KB
    __bf16* Wh_bf  = (__bf16*)ws;                     ws += (size_t)HH * HH * 2;  // 512 KB
    __bf16* Wo_bf  = (__bf16*)ws;                     /* 1 MB */                  // ~16 MB total

    // ---- bf16 packing (memory-bound, L2-resident) ----
    const int nQ = BT * HH, nW = HH * HH, nWo = HH * 2 * HH;
    pack_bf16_kernel<<<dim3(nQ / 1024),  dim3(256), 0, stream>>>(query, q_bf, nQ);
    pack_bf16_kernel<<<dim3(nQ / 1024),  dim3(256), 0, stream>>>(enc,   e_bf, nQ);
    pack_bf16_kernel<<<dim3(nW / 1024),  dim3(256), 0, stream>>>(W_s,   Ws_bf, nW);
    pack_bf16_kernel<<<dim3(nW / 1024),  dim3(256), 0, stream>>>(W_h,   Wh_bf, nW);
    pack_bf16_kernel<<<dim3(nWo / 1024), dim3(256), 0, stream>>>(W_out, Wo_bf, nWo);

    // ---- projections: Ws_q = q @ W_s^T ; Wh_e = enc @ W_h^T ----
    dim3 blk(128);
    dim3 gProj(BT / 16, HH / 256);                    // (128, 2)
    gemm_bf16_kernel<<<gProj, blk, 0, stream>>>(q_bf, Ws_bf, wsq, HH, HH);
    gemm_bf16_kernel<<<gProj, blk, 0, stream>>>(e_bf, Wh_bf, whe, HH, HH);

    // ---- fused energies + mask + softmax + context ----
    energies_softmax_ctx_kernel<<<dim3(BB * TT), dim3(256), 0, stream>>>(
        wsq, whe, enc, v, lens, ctx_bf);

    // ---- attn_out = tanh([ctx|query] @ W_out^T) ----
    gemm_out_kernel<<<gProj, blk, 0, stream>>>(ctx_bf, q_bf, Wo_bf, out);
}
